// GeneralSampleEdgeConv_19731079758632
// MI455X (gfx1250) — matrix operand order repaired and checked by reading, not compiled, not run on hardware
//
#include <hip/hip_runtime.h>

typedef __attribute__((ext_vector_type(16))) __bf16 v16bf;
typedef __attribute__((ext_vector_type(8)))  __bf16 v8bf;
typedef __attribute__((ext_vector_type(8)))  float  v8f;
typedef __attribute__((ext_vector_type(4)))  float  f32x4;

#define D_FEAT 96
#define K_DIM  192
#define W_STRIDE 200   // padded bf16 row stride for the transposed W in LDS (bank-conflict-free b128 reads)

__global__ void zero_out_kernel(float* __restrict__ out, int n) {
    int i = blockIdx.x * blockDim.x + threadIdx.x;
    int stride = gridDim.x * blockDim.x;
    for (; i < n; i += stride) out[i] = 0.0f;
}

__global__ __launch_bounds__(256) void edgeconv_wmma_kernel(
    const float* __restrict__ node_feature,   // [N, 96]
    const float* __restrict__ edge_feature,   // [E, 96]
    const int*   __restrict__ edge_src,       // [E]
    const int*   __restrict__ edge_dst,       // [E]
    const unsigned char* __restrict__ edge_mask, // [E] bool bytes
    const float* __restrict__ W,              // [192, 96] row-major
    float* __restrict__ out,                  // [N, 96]
    int E, int numTiles)
{
    // Stage W transposed as bf16: Wlds[n * W_STRIDE + k] = bf16(W[k,n])
    __shared__ __bf16 Wlds[D_FEAT * W_STRIDE];
    const int tid = threadIdx.x;
    for (int i = tid; i < K_DIM * D_FEAT; i += 256) {
        int k = i / D_FEAT;
        int n = i % D_FEAT;
        Wlds[n * W_STRIDE + k] = (__bf16)W[i];
    }
    __syncthreads();

    const int lane = tid & 31;
    const int wid  = tid >> 5;
    const int half = lane >> 4;     // 0: lanes 0-15, 1: lanes 16-31
    const int m16  = lane & 15;
    const int wavesTotal = gridDim.x * (blockDim.x >> 5);

    for (int tile = blockIdx.x * (blockDim.x >> 5) + wid; tile < numTiles; tile += wavesTotal) {
        // This lane supplies A-matrix row M = m16 of the 16-edge tile.
        int eRow = tile * 16 + m16;
        int srow = 0;
        float scale = 0.0f;
        if (eRow < E) {
            srow  = edge_src[eRow];
            scale = edge_mask[eRow] ? 1.0f : 0.0f;   // mask folded into inputs (linear op)
        }
        const float* nodeRow = node_feature + (size_t)srow * D_FEAT;
        const float* edgeRow = edge_feature + (size_t)(eRow < E ? eRow : 0) * D_FEAT;

        // Hoist dst index loads so they overlap the WMMA sequence.
        // C/D layout: VGPR v -> M = v + 8*half, N = m16.
        int dIdx[8];
#pragma unroll
        for (int v = 0; v < 8; ++v) {
            int e = tile * 16 + half * 8 + v;
            dIdx[v] = (e < E) ? edge_dst[e] : -1;
        }

        v8f acc[6];
#pragma unroll
        for (int nt = 0; nt < 6; ++nt) acc[nt] = (v8f){0.f,0.f,0.f,0.f,0.f,0.f,0.f,0.f};

#pragma unroll
        for (int kc = 0; kc < 6; ++kc) {
            // ---- A fragment (16x32 bf16 layout): lanes hold K = k0..k0+15
            //      with k0 = kc*32 (+16 for lanes 16-31).
            const int k0 = kc * 32 + (half ? 16 : 0);
            const float* base;
            int col;
            if (kc < 3) { base = nodeRow; col = k0; }           // concat cols 0..95  -> node feature
            else        { base = edgeRow; col = k0 - D_FEAT; }  // concat cols 96..191 -> edge feature

            f32x4 f0 = *(const f32x4*)(base + col);
            f32x4 f1 = *(const f32x4*)(base + col + 4);
            f32x4 f2 = *(const f32x4*)(base + col + 8);
            f32x4 f3 = *(const f32x4*)(base + col + 12);

            v16bf a;
#pragma unroll
            for (int j = 0; j < 4; ++j) {
                a[j]      = (__bf16)(f0[j] * scale);
                a[4 + j]  = (__bf16)(f1[j] * scale);
                a[8 + j]  = (__bf16)(f2[j] * scale);
                a[12 + j] = (__bf16)(f3[j] * scale);
            }

            // ---- B fragments (32x16 bf16 layout): lane holds column N = nt*16 + m16,
            //      K = kc*32 + (half ? 16 : 0) .. +15, i.e. 16 contiguous bf16 in transposed LDS.
            const int kb = kc * 32 + (half ? 16 : 0);
#pragma unroll
            for (int nt = 0; nt < 6; ++nt) {
                const int colN = nt * 16 + m16;
                const v8bf* bp = (const v8bf*)&Wlds[colN * W_STRIDE + kb];
                v8bf blo = bp[0];
                v8bf bhi = bp[1];
                v16bf b = __builtin_shufflevector(blo, bhi,
                                                  0,1,2,3,4,5,6,7,8,9,10,11,12,13,14,15);
                acc[nt] = __builtin_amdgcn_wmma_f32_16x16x32_bf16(
                    /*neg_a=*/false, a, /*neg_b=*/false, b,
                    /*c_mod=*/(short)0, acc[nt], /*reuse_a=*/false, /*reuse_b=*/false);
            }
        }

        // ---- Scatter-add with hardware fp32 atomics (no-return global_atomic_add_f32).
#pragma unroll
        for (int nt = 0; nt < 6; ++nt) {
#pragma unroll
            for (int v = 0; v < 8; ++v) {
                if (dIdx[v] >= 0) {
                    unsafeAtomicAdd(out + (size_t)dIdx[v] * D_FEAT + nt * 16 + m16, acc[nt][v]);
                }
            }
        }
    }
}

extern "C" void kernel_launch(void* const* d_in, const int* in_sizes, int n_in,
                              void* d_out, int out_size, void* d_ws, size_t ws_size,
                              hipStream_t stream) {
    const float* node_feature = (const float*)d_in[0];
    const float* edge_feature = (const float*)d_in[1];
    const int*   edge_index   = (const int*)d_in[2];   // [2, E]: row 0 = src, row 1 = dst
    const unsigned char* edge_mask = (const unsigned char*)d_in[3];
    const float* W_msg        = (const float*)d_in[4];
    float* out = (float*)d_out;

    const int E = in_sizes[3];
    const int numTiles = (E + 15) / 16;

    zero_out_kernel<<<1024, 256, 0, stream>>>(out, out_size);

    const int blocks = 640;  // 5120 waves; ~10 tiles/wave; W staging amortized via L2
    edgeconv_wmma_kernel<<<blocks, 256, 0, stream>>>(
        node_feature, edge_feature,
        edge_index, edge_index + E, edge_mask, W_msg,
        out, E, numTiles);
}
